// ActorCritic_68951404970626
// MI455X (gfx1250) — matrix operand order repaired and checked by reading, not compiled, not run on hardware
//
#include <hip/hip_runtime.h>
#include <hip/hip_bf16.h>

// ---------------------------------------------------------------------------
// Problem constants
// ---------------------------------------------------------------------------
#define B_   64
#define S_   1024
#define E_   512
#define A_   128
#define SM1  1023          // S-1 score positions
#define K2E  1024          // 2*E, K dimension of the big GEMMs

typedef __attribute__((ext_vector_type(16))) __bf16 v16bf;
typedef __attribute__((ext_vector_type(8)))  __bf16 v8bf;
typedef __attribute__((ext_vector_type(8)))  float  v8f;

__device__ __forceinline__ __bf16 f2bf(float f) {
    unsigned u = __builtin_bit_cast(unsigned, f);
    unsigned r = u + 0x7FFFu + ((u >> 16) & 1u);   // round-to-nearest-even
    unsigned short h = (unsigned short)(r >> 16);
    return __builtin_bit_cast(__bf16, h);
}

__device__ __forceinline__ v16bf cat16(v8bf lo, v8bf hi) {
    return __builtin_shufflevector(lo, hi, 0,1,2,3,4,5,6,7,8,9,10,11,12,13,14,15);
}

// ---------------------------------------------------------------------------
// Prep kernel: W (1024x1024 f32, k-major) -> WT (1024x1024 bf16, n-major).
// Tiled 64x64 transpose through LDS so both global sides stay coalesced.
// ---------------------------------------------------------------------------
__global__ void __launch_bounds__(256)
transpose_bf16_kernel(const float* __restrict__ src, __bf16* __restrict__ dst)
{
    __shared__ __bf16 tl[64][72];          // padded to dodge bank conflicts
    const int k0  = blockIdx.x * 64;
    const int n0  = blockIdx.y * 64;
    const int tid = threadIdx.x;
    for (int idx = tid; idx < 4096; idx += 256) {
        int r = idx >> 6;                  // k offset
        int c = idx & 63;                  // n offset (coalesced)
        tl[c][r] = f2bf(src[(size_t)(k0 + r) * K2E + n0 + c]);
    }
    __syncthreads();
    for (int idx = tid; idx < 4096; idx += 256) {
        int r = idx >> 6;                  // n offset
        int c = idx & 63;                  // k offset (coalesced)
        dst[(size_t)(n0 + r) * K2E + k0 + c] = tl[r][c];
    }
}

// ---------------------------------------------------------------------------
// Kernel 1: fused position head.
//   scores[b,s] = sum_n relu( concat(st[b,s],st[b,s+1]) . W1p[:,n] + b1p[n] ) * w2p[n]
// One block = 128 rows of one batch, 8 waves, 16-row WMMA M tile per wave.
// LDS: 129x512 bf16 states window (129KB) + double-buffered 16x1024 bf16
// W1p panels (2x32KB) filled with global_load_async_to_lds_b128 (ASYNCcnt).
// ---------------------------------------------------------------------------
#define XS_ELEMS  (129 * 512)
#define XS_BYTES  (XS_ELEMS * 2)                 // 132096
#define BS_BYTES  (16 * 1024 * 2)                // 32768 per panel
#define K1_LDS    (XS_BYTES + 2 * BS_BYTES)      // 197632 (<= 320KB/WGP)

__global__ void __launch_bounds__(256)
pos_scores_kernel(const float* __restrict__ states,
                  const __bf16* __restrict__ W1pT,   // [n][k] bf16, n-major
                  const float* __restrict__ b1p,
                  const float* __restrict__ w2p,
                  float* __restrict__ scores)
{
    extern __shared__ __align__(16) char dynsmem[];   // LDS offset 0 (no static LDS)
    __bf16* Xs  = (__bf16*)dynsmem;                   // [129][512]
    __bf16* BsA = (__bf16*)(dynsmem + XS_BYTES);      // panel buffer 0
    __bf16* BsB = (__bf16*)(dynsmem + XS_BYTES + BS_BYTES); // panel buffer 1

    const int blk  = blockIdx.x;
    const int b    = blk >> 3;             // batch
    const int t    = blk & 7;              // row tile within batch
    const int s0   = t * 128;
    const int tid  = threadIdx.x;
    const int lane = tid & 31;
    const int w    = tid >> 5;             // wave id 0..7
    const int half = lane >> 4;
    const int ml   = lane & 15;            // M for A-frag, N for B/C-frag
    const int m0   = w * 16;               // wave's M offset within block tile

    // async fetch of one contiguous 32KB bf16 panel: W1pT[n0:n0+16][:]
    auto issue_panel = [&](int nt, int buf) {
        const unsigned long long gbase =
            (unsigned long long)(uintptr_t)(W1pT + (size_t)nt * 16 * K2E);
        unsigned lds  = (unsigned)(XS_BYTES + buf * BS_BYTES + tid * 16);
        unsigned voff = (unsigned)(tid * 16);
        #pragma unroll
        for (int i = 0; i < 8; ++i) {
            asm volatile("global_load_async_to_lds_b128 %0, %1, %2"
                         :: "v"(lds), "v"(voff), "s"(gbase) : "memory");
            lds  += 4096;
            voff += 4096;
        }
    };

    // ---- stage 129x512 f32 window -> bf16 LDS (zero-fill past S) ----------
    {
        const float4* src4 = (const float4*)(states + ((size_t)b * S_ + s0) * E_);
        for (int i = tid; i < 129 * 128; i += 256) {
            int r  = i >> 7;               // window row 0..128
            int c4 = i & 127;              // float4 column
            float4 v = make_float4(0.f, 0.f, 0.f, 0.f);
            if (s0 + r < S_) v = src4[(size_t)r * 128 + c4];
            __bf16* dst = Xs + (size_t)r * 512 + c4 * 4;
            dst[0] = f2bf(v.x); dst[1] = f2bf(v.y);
            dst[2] = f2bf(v.z); dst[3] = f2bf(v.w);
        }
    }
    issue_panel(0, 0);                     // preload first panel

    float sp[8];
    #pragma unroll
    for (int v = 0; v < 8; ++v) sp[v] = 0.f;

    // ---- loop over 64 N tiles of 16 features ------------------------------
    for (int nt = 0; nt < 64; ++nt) {
        const int n0 = nt * 16;

        asm volatile("s_wait_asynccnt 0" ::: "memory"); // my async copies landed
        __syncthreads();                                // everyone's copies landed,
                                                        // previous buffer reads done
        if (nt + 1 < 64) issue_panel(nt + 1, (nt + 1) & 1);

        const __bf16* Bs = (nt & 1) ? BsB : BsA;

        v8f acc = {0.f, 0.f, 0.f, 0.f, 0.f, 0.f, 0.f, 0.f};

        // ---- K loop: 32 x v_wmma_f32_16x16x32_bf16 -----------------------
        #pragma unroll 4
        for (int kt = 0; kt < 32; ++kt) {
            const int kbase = kt * 32;
            const int koff  = (kbase < 512) ? 0 : 1;   // 2nd K half uses row s+1
            const int kb    = kbase & 511;
            const int r     = m0 + ml + koff;

            // A fragment (16x32 bf16): lane 16*half+M holds K = half*8+{0..7}
            // and 16+half*8+{0..7} of row M  -> two 16B LDS loads
            const __bf16* arow = Xs + (size_t)r * 512 + kb + half * 8;
            v8bf a0 = *(const v8bf*)(arow);
            v8bf a1 = *(const v8bf*)(arow + 16);
            v16bf Af = cat16(a0, a1);

            // B fragment (32x16 bf16): lane 16*half+N holds K = half*8+{0..7}
            // and 16+half*8+{0..7}
            const __bf16* brow = Bs + (size_t)ml * K2E + kbase + half * 8;
            v8bf b0 = *(const v8bf*)(brow);
            v8bf b1 = *(const v8bf*)(brow + 16);
            v16bf Bf = cat16(b0, b1);

            acc = __builtin_amdgcn_wmma_f32_16x16x32_bf16(
                      false, Af, false, Bf, (short)0, acc, false, false);
        }

        // epilogue: bias + relu + dot with w2p slice, fused into score partial
        const float bias = b1p[n0 + ml];
        const float wv   = w2p[n0 + ml];
        #pragma unroll
        for (int v = 0; v < 8; ++v) {
            float h = acc[v] + bias;
            h = fmaxf(h, 0.f);
            sp[v] += h * wv;
        }
    }

    // ---- reduce the 16 N-lane partials inside each half, then store -------
    #pragma unroll
    for (int v = 0; v < 8; ++v) {
        float x = sp[v];
        #pragma unroll
        for (int off = 8; off >= 1; off >>= 1)
            x += __shfl_xor(x, off, 16);
        sp[v] = x;
    }
    if (ml == 0) {
        #pragma unroll
        for (int v = 0; v < 8; ++v) {
            int srow = s0 + m0 + v + 8 * half;
            if (srow < SM1)
                scores[(size_t)b * SM1 + srow] = sp[v];   // b2p cancels in softmax
        }
    }
}

// ---------------------------------------------------------------------------
// Kernel 2: masked log-softmax over positions + entropy.  1 block / batch.
// ---------------------------------------------------------------------------
__global__ void __launch_bounds__(256)
pos_softmax_kernel(const float* __restrict__ scores,
                   const int* __restrict__ lengths,
                   const int* __restrict__ pos_action,
                   float* __restrict__ out)
{
    __shared__ float red[256];
    __shared__ float sM, sSE, sSS;
    const int b   = blockIdx.x;
    const int tid = threadIdx.x;
    const int nv  = lengths[b] - 1;
    const float* sc = scores + (size_t)b * SM1;

    float mx = -3.402823466e38f;
    for (int i = tid; i < SM1; i += 256)
        if (i < nv) mx = fmaxf(mx, sc[i]);
    red[tid] = mx; __syncthreads();
    for (int o = 128; o >= 1; o >>= 1) {
        if (tid < o) red[tid] = fmaxf(red[tid], red[tid + o]);
        __syncthreads();
    }
    if (tid == 0) sM = red[0];
    __syncthreads();
    const float M = sM;

    float se = 0.f, ss = 0.f;
    for (int i = tid; i < SM1; i += 256) {
        if (i < nv) {
            float s = sc[i];
            float e = __expf(s - M);
            se += e; ss += e * s;
        }
    }
    red[tid] = se; __syncthreads();
    for (int o = 128; o >= 1; o >>= 1) {
        if (tid < o) red[tid] += red[tid + o];
        __syncthreads();
    }
    if (tid == 0) sSE = red[0];
    __syncthreads();
    red[tid] = ss; __syncthreads();
    for (int o = 128; o >= 1; o >>= 1) {
        if (tid < o) red[tid] += red[tid + o];
        __syncthreads();
    }
    if (tid == 0) sSS = red[0];
    __syncthreads();

    if (tid == 0) {
        float L = __logf(sSE);
        int pa  = pos_action[b];
        out[0 * B_ + b] = sc[pa] - M - L;          // logp_pos
        out[3 * B_ + b] = (M + L) - sSS / sSE;     // ent_pos
    }
}

// ---------------------------------------------------------------------------
// Kernel 3a: symbol head GEMM1 with WMMA, B straight from transposed bf16
// weights (two global_load_b128 per fragment).
// 4 blocks (16 batch rows each), 8 waves; wave w owns N tiles w*8..w*8+7.
// ---------------------------------------------------------------------------
__global__ void __launch_bounds__(256)
sym_gemm1_kernel(const float* __restrict__ states,
                 const int* __restrict__ pos_action,
                 const __bf16* __restrict__ W1sT,   // [n][k] bf16, n-major
                 const float* __restrict__ b1s,
                 float* __restrict__ sh)
{
    __shared__ __bf16 Xs[16 * 1024];     // 32 KB
    const int tb   = blockIdx.x;         // 0..3
    const int tid  = threadIdx.x;
    const int lane = tid & 31;
    const int w    = tid >> 5;
    const int half = lane >> 4;
    const int ml   = lane & 15;

    // gather 16 rows of concat(e1,e2) -> bf16 LDS
    for (int i = tid; i < 16 * 256; i += 256) {
        int r  = i >> 8;                 // local row
        int c4 = i & 255;                // float4 col (k = c4*4)
        int bb = tb * 16 + r;
        int pa = pos_action[bb];
        int k  = c4 * 4;
        const float4* src;
        if (k < 512)
            src = (const float4*)(states + ((size_t)bb * S_ + pa) * E_ + k);
        else
            src = (const float4*)(states + ((size_t)bb * S_ + pa + 1) * E_ + (k - 512));
        float4 v = *src;
        __bf16* dst = Xs + (size_t)r * 1024 + k;
        dst[0] = f2bf(v.x); dst[1] = f2bf(v.y);
        dst[2] = f2bf(v.z); dst[3] = f2bf(v.w);
    }
    __syncthreads();

    for (int j = 0; j < 8; ++j) {
        const int n0 = (w * 8 + j) * 16;
        v8f acc = {0.f, 0.f, 0.f, 0.f, 0.f, 0.f, 0.f, 0.f};
        #pragma unroll 4
        for (int kt = 0; kt < 32; ++kt) {
            const int kbase = kt * 32;
            const __bf16* arow = Xs + (size_t)ml * 1024 + kbase + half * 8;
            v8bf a0 = *(const v8bf*)(arow);
            v8bf a1 = *(const v8bf*)(arow + 16);
            v16bf Af = cat16(a0, a1);

            const __bf16* brow = W1sT + (size_t)(n0 + ml) * K2E + kbase + half * 8;
            v8bf b0 = *(const v8bf*)(brow);
            v8bf b1 = *(const v8bf*)(brow + 16);
            v16bf Bf = cat16(b0, b1);

            acc = __builtin_amdgcn_wmma_f32_16x16x32_bf16(
                      false, Af, false, Bf, (short)0, acc, false, false);
        }
        const float bias = b1s[n0 + ml];
        #pragma unroll
        for (int v = 0; v < 8; ++v) {
            int row = v + 8 * half;
            float hv = fmaxf(acc[v] + bias, 0.f);
            sh[(size_t)(tb * 16 + row) * K2E + n0 + ml] = hv;
        }
    }
}

// ---------------------------------------------------------------------------
// Kernel 3b: logits = sh @ W2s + b2s, then log-softmax over A=128 + entropy.
// ---------------------------------------------------------------------------
__global__ void __launch_bounds__(128)
sym_logits_kernel(const float* __restrict__ sh,
                  const float* __restrict__ W2s,
                  const float* __restrict__ b2s,
                  const int* __restrict__ sym_action,
                  float* __restrict__ out)
{
    __shared__ float row[1024];
    __shared__ float red[128];
    __shared__ float sM, sSE, sSS, sSel;
    const int b   = blockIdx.x;
    const int tid = threadIdx.x;

    for (int i = tid; i < K2E; i += 128) row[i] = sh[(size_t)b * K2E + i];
    __syncthreads();

    float lg = b2s[tid];
    for (int k = 0; k < K2E; ++k)
        lg = fmaf(row[k], W2s[(size_t)k * A_ + tid], lg);

    red[tid] = lg; __syncthreads();
    for (int o = 64; o >= 1; o >>= 1) {
        if (tid < o) red[tid] = fmaxf(red[tid], red[tid + o]);
        __syncthreads();
    }
    if (tid == 0) sM = red[0];
    __syncthreads();
    const float M = sM;
    float e = __expf(lg - M);
    red[tid] = e; __syncthreads();
    for (int o = 64; o >= 1; o >>= 1) {
        if (tid < o) red[tid] += red[tid + o];
        __syncthreads();
    }
    if (tid == 0) sSE = red[0];
    __syncthreads();
    red[tid] = e * lg; __syncthreads();
    for (int o = 64; o >= 1; o >>= 1) {
        if (tid < o) red[tid] += red[tid + o];
        __syncthreads();
    }
    if (tid == 0) sSS = red[0];
    if (tid == sym_action[b]) sSel = lg;
    __syncthreads();

    if (tid == 0) {
        float L = __logf(sSE);
        out[1 * B_ + b] = sSel - M - L;            // logp_sym
        out[4 * B_ + b] = (M + L) - sSS / sSE;     // ent_sym
    }
}

// ---------------------------------------------------------------------------
// Kernel 4: value head.  val = relu(cls @ Wc1 + bc1) . wc2 + bc2
// ---------------------------------------------------------------------------
__global__ void __launch_bounds__(256)
value_kernel(const float* __restrict__ cls,
             const float* __restrict__ Wc1,
             const float* __restrict__ bc1,
             const float* __restrict__ wc2,
             const float* __restrict__ bc2,
             float* __restrict__ out)
{
    __shared__ float crow[E_];
    __shared__ float red[256];
    const int b   = blockIdx.x;
    const int tid = threadIdx.x;

    for (int i = tid; i < E_; i += 256) crow[i] = cls[(size_t)b * E_ + i];
    __syncthreads();

    float acc = 0.f;
    #pragma unroll
    for (int half = 0; half < 2; ++half) {
        int f = tid + half * 256;
        float h = bc1[f];
        for (int k = 0; k < E_; ++k)
            h = fmaf(crow[k], Wc1[(size_t)k * E_ + f], h);
        acc += fmaxf(h, 0.f) * wc2[f];
    }
    red[tid] = acc; __syncthreads();
    for (int o = 128; o >= 1; o >>= 1) {
        if (tid < o) red[tid] += red[tid + o];
        __syncthreads();
    }
    if (tid == 0) out[2 * B_ + b] = red[0] + bc2[0];
}

// ---------------------------------------------------------------------------
// Launch
// ---------------------------------------------------------------------------
extern "C" void kernel_launch(void* const* d_in, const int* in_sizes, int n_in,
                              void* d_out, int out_size, void* d_ws, size_t ws_size,
                              hipStream_t stream)
{
    (void)in_sizes; (void)n_in; (void)out_size; (void)ws_size;
    const float* states = (const float*)d_in[0];
    const float* cls    = (const float*)d_in[1];
    const float* W1p    = (const float*)d_in[2];
    const float* b1p    = (const float*)d_in[3];
    const float* w2p    = (const float*)d_in[4];
    /* b2p (d_in[5]) cancels in log-softmax */
    const float* W1s    = (const float*)d_in[6];
    const float* b1s    = (const float*)d_in[7];
    const float* W2s    = (const float*)d_in[8];
    const float* b2s    = (const float*)d_in[9];
    const float* Wc1    = (const float*)d_in[10];
    const float* bc1    = (const float*)d_in[11];
    const float* wc2    = (const float*)d_in[12];
    const float* bc2    = (const float*)d_in[13];
    const int* lengths  = (const int*)d_in[14];
    const int* pos_act  = (const int*)d_in[15];
    const int* sym_act  = (const int*)d_in[16];
    float* out = (float*)d_out;

    // workspace layout
    __bf16* W1pT  = (__bf16*)d_ws;                              // 2 MB
    __bf16* W1sT  = (__bf16*)((char*)d_ws + 2097152);           // 2 MB
    float*  scores = (float*)((char*)d_ws + 4194304);           // 64*1023 f32
    float*  sh     = (float*)((char*)d_ws + 4194304 + 262144);  // 64*1024 f32

    transpose_bf16_kernel<<<dim3(16, 16), dim3(256), 0, stream>>>(W1p, W1pT);
    transpose_bf16_kernel<<<dim3(16, 16), dim3(256), 0, stream>>>(W1s, W1sT);

    pos_scores_kernel<<<dim3(B_ * 8), dim3(256), K1_LDS, stream>>>(
        states, W1pT, b1p, w2p, scores);
    pos_softmax_kernel<<<dim3(B_), dim3(256), 0, stream>>>(
        scores, lengths, pos_act, out);
    sym_gemm1_kernel<<<dim3(4), dim3(256), 0, stream>>>(
        states, pos_act, W1sT, b1s, sh);
    sym_logits_kernel<<<dim3(B_), dim3(128), 0, stream>>>(
        sh, W2s, b2s, sym_act, out);
    value_kernel<<<dim3(B_), dim3(256), 0, stream>>>(
        cls, Wc1, bc1, wc2, bc2, out);
}
